// AudioConv2DSelfAttentionBlock_42502996361795
// MI455X (gfx1250) — matrix operand order repaired and checked by reading, not compiled, not run on hardware
//
#include <hip/hip_runtime.h>
#include <hip/hip_bf16.h>

// ---------------------------------------------------------------------------
// AudioConv2DSelfAttentionBlock for MI455X (gfx1250), bf16 WMMA pipeline.
// B=4 C=256 M=16 T=128 (S=2048) H=8 DQ=DV=64, score scale 1/8 folded into q.
// V stored transposed [b][h][d][s] so PV B-fragments load contiguously.
// ---------------------------------------------------------------------------

typedef __attribute__((ext_vector_type(16))) __bf16 v16bf;
typedef __attribute__((ext_vector_type(8)))  float  v8f;

#define Bsz 4
#define Csz 256
#define Msz 16
#define Tsz 128
#define Ssz 2048     // M*T
#define Hsz 8
#define Dsz 64
#define Osz 512      // H*DQ = H*DV

// float -> bf16 with round-to-nearest-even (avoids relying on fptrunc-to-bf16)
static __device__ inline __bf16 f2bf(float f) {
    union { float f; unsigned u; } v; v.f = f;
    unsigned r = v.u + 0x7FFFu + ((v.u >> 16) & 1u);
    unsigned short h = (unsigned short)(r >> 16);
    return __builtin_bit_cast(__bf16, h);
}

// ---------------------------------------------------------------------------
// RoPE angle tables: cos/sin, shape [S][32].
// cols 0..15: freq position m (period-8 repeat); cols 16..31: time position t.
// ---------------------------------------------------------------------------
__global__ void k_rope_table(float* __restrict__ cosT, float* __restrict__ sinT) {
    int idx = blockIdx.x * blockDim.x + threadIdx.x;
    if (idx >= Ssz * 32) return;
    int j = idx & 31;
    int s = idx >> 5;
    int m = s >> 7, t = s & 127;
    float pos = (j < 16) ? (float)m : (float)t;
    int jj = j & 7;                           // inv-freq index (repeats with period 8)
    float inv = powf(10000.0f, -(float)(2 * jj) / 16.0f);
    float a = pos * inv;
    cosT[idx] = cosf(a);
    sinT[idx] = sinf(a);
}

__global__ void k_cvt_bf16(const float* __restrict__ src, __bf16* __restrict__ dst, int n) {
    int i = blockIdx.x * blockDim.x + threadIdx.x;
    if (i < n) dst[i] = f2bf(src[i]);
}

// ---------------------------------------------------------------------------
// Depthwise 3x3 conv (SAME, cross-correlation) + bias -> bf16 [b][s][c]
// One block per (b, c) plane: reads fully coalesced; scattered 2B writes are
// absorbed by the 192MB L2 (total write footprint ~12.6MB).
// ---------------------------------------------------------------------------
__global__ void k_dwconv(const float* __restrict__ x, const float* __restrict__ w,
                         const float* __restrict__ bias, __bf16* __restrict__ dst) {
    int b = blockIdx.x >> 8;
    int c = blockIdx.x & 255;
    const float* xp = x + ((size_t)b * Csz + c) * Ssz;
    float wr[9];
#pragma unroll
    for (int i = 0; i < 9; i++) wr[i] = w[c * 9 + i];
    float bv = bias[c];
    for (int idx = threadIdx.x; idx < Ssz; idx += blockDim.x) {
        int m = idx >> 7, t = idx & 127;
        float acc = bv;
#pragma unroll
        for (int i = 0; i < 3; i++) {
            int mm = m + i - 1;
            if (mm < 0 || mm >= Msz) continue;
#pragma unroll
            for (int j = 0; j < 3; j++) {
                int tt = t + j - 1;
                if (tt < 0 || tt >= Tsz) continue;
                acc += xp[mm * Tsz + tt] * wr[i * 3 + j];
            }
        }
        dst[((size_t)b * Ssz + idx) * Csz + c] = f2bf(acc);
    }
}

// ---------------------------------------------------------------------------
// Pointwise conv as GEMM: Y[b,s,o] = sum_c A[b,s,c] * W[o,c] + bias[o]
// A bf16 [b][s][C], W bf16 [O][C]. 256-thr blocks = 8 waves = 8 16x16 tiles.
// ---------------------------------------------------------------------------
__global__ void k_pwgemm(const __bf16* __restrict__ A, const __bf16* __restrict__ W,
                         const float* __restrict__ bias, float* __restrict__ Y) {
    int tile = blockIdx.x * 8 + (threadIdx.x >> 5);
    int ot = tile & 31;  tile >>= 5;   // O/16 = 32 tiles
    int st = tile & 127; tile >>= 7;   // S/16 = 128 tiles
    int b  = tile;
    int lane = threadIdx.x & 31;
    int hi = lane >> 4;
    int khalfA = hi << 3;              // unused; A also uses kbaseB
    int kbaseB = hi << 4;              // B element->K half offset
    int mrow = lane & 15;
    int n = lane & 15;
    int s0 = st * 16, o0 = ot * 16;
    v8f acc = {};
    const __bf16* arow = A + ((size_t)b * Ssz + (s0 + mrow)) * Csz;
    const __bf16* brow = W + (size_t)(o0 + n) * Csz;
    for (int k0 = 0; k0 < Csz; k0 += 32) {
        v16bf a, bb;
#pragma unroll
        for (int e = 0; e < 16; e++) {
            int ka = kbaseB + e;
            a[e]  = arow[k0 + ka];
            bb[e] = brow[k0 + kbaseB + e];
        }
        acc = __builtin_amdgcn_wmma_f32_16x16x32_bf16(false, a, false, bb,
                                                      (short)0, acc, false, false);
    }
    float bv = bias[o0 + n];
#pragma unroll
    for (int r = 0; r < 8; r++) {
        int row = (hi << 3) + r;
        Y[((size_t)b * Ssz + s0 + row) * Osz + o0 + n] = acc[r] + bv;
    }
}

// ---------------------------------------------------------------------------
// RoPE apply + layout change. Y fp32 [3][B][S][512] ->
//   qb, kb bf16 [b*H+h][s][64]   (q gets 1/sqrt(DQ)=0.125 folded in)
//   vt     bf16 [b*H+h][d][s]    (transposed, for contiguous PV B-fragments)
// ---------------------------------------------------------------------------
__global__ void k_rope_apply(const float* __restrict__ Y,
                             const float* __restrict__ cosT, const float* __restrict__ sinT,
                             __bf16* __restrict__ qb, __bf16* __restrict__ kb,
                             __bf16* __restrict__ vt) {
    int idx = blockIdx.x * blockDim.x + threadIdx.x;      // ((b*8+h)*2048+s)*32+d
    if (idx >= Bsz * Hsz * Ssz * 32) return;
    int d = idx & 31;
    int s = (idx >> 5) & (Ssz - 1);
    int h = (idx >> 16) & 7;
    int b = idx >> 19;
    size_t ybase = ((size_t)b * Ssz + s) * Osz + h * Dsz;
    const float* yq = Y + ybase;
    const float* yk = Y + (size_t)Bsz * Ssz * Osz + ybase;
    const float* yv = Y + 2 * (size_t)Bsz * Ssz * Osz + ybase;
    float c = cosT[s * 32 + d], sn = sinT[s * 32 + d];
    size_t bh = (size_t)(b * Hsz + h);
    size_t obase = (bh * Ssz + s) * Dsz;
    float q1 = yq[d], q2 = yq[d + 32];
    qb[obase + d]      = f2bf(0.125f * (q1 * c - q2 * sn));
    qb[obase + d + 32] = f2bf(0.125f * (q1 * sn + q2 * c));
    float k1 = yk[d], k2 = yk[d + 32];
    kb[obase + d]      = f2bf(k1 * c - k2 * sn);
    kb[obase + d + 32] = f2bf(k1 * sn + k2 * c);
    size_t vtbase = bh * Dsz * Ssz + s;                   // [bh][d][s]
    vt[vtbase + (size_t)d * Ssz]        = f2bf(yv[d]);
    vt[vtbase + (size_t)(d + 32) * Ssz] = f2bf(yv[d + 32]);
}

// ---------------------------------------------------------------------------
// Flash attention. One workgroup (4 waves / 128 thr) per (b, h, 16-row q tile).
// Each wave: one 16-col score tile per 64-col chunk, one 16-wide d slice of O.
// P staged via LDS (bf16) so PV A-fragments match the ISA A layout.
// Softmax stats parallelized: 128 threads = 16 rows x 8 column segments.
// ---------------------------------------------------------------------------
__global__ void k_attn(const __bf16* __restrict__ qb, const __bf16* __restrict__ kb,
                       const __bf16* __restrict__ vt, const unsigned char* __restrict__ mask,
                       __bf16* __restrict__ ob) {
    __shared__ float  sS[16][64];
    __shared__ __bf16 sP[16][64];
    __shared__ float  sRed[16][8];
    __shared__ float  rowM[16], rowL[16], rowFac[16];

    int bid = blockIdx.x;
    int qt = bid & 127; bid >>= 7;
    int h  = bid & 7;   bid >>= 3;
    int b  = bid;
    int tid  = threadIdx.x;
    int wave = tid >> 5;
    int lane = tid & 31;
    int hi = lane >> 4;
    int khalfA = hi << 3;
    int kbaseB = hi << 4;
    int mrow = lane & 15;
    int n = lane & 15;
    int sq0 = qt * 16;
    int dwave = wave * 16;
    size_t bh = (size_t)(b * Hsz + h) * Ssz;
    int row8 = tid >> 3;               // 0..15  (stats row)
    int seg  = tid & 7;                // 0..7   (stats 8-col segment)

    // preload q A-fragments (K = d in [0,32) and [32,64))
    v16bf aq0, aq1;
    const __bf16* qrow = qb + (bh + sq0 + mrow) * Dsz;
#pragma unroll
    for (int e = 0; e < 16; e++) {
        int ka = kbaseB + e;
        aq0[e] = qrow[ka];
        aq1[e] = qrow[32 + ka];
    }
    // V row for this wave's d-slice / lane column (transposed layout)
    const __bf16* vrow = vt + ((bh * Dsz) + (size_t)(dwave + n) * Ssz);
    if (tid < 16) { rowM[tid] = -3.0e38f; rowL[tid] = 0.0f; }
    v8f accO = {};

    for (int chunk = 0; chunk < Ssz / 64; chunk++) {
        int sk0 = chunk * 64 + wave * 16;
        const __bf16* krow = kb + (bh + sk0 + n) * Dsz + kbaseB;
        if (chunk + 1 < Ssz / 64) {     // prefetch next chunk's K and V tiles
            __builtin_prefetch(krow + 64 * Dsz, 0, 1);
            __builtin_prefetch(vrow + (chunk + 1) * 64, 0, 1);
        }
        __syncthreads();
        // ---- scores for this wave's 16 key columns ----
        v8f sc = {};
        v16bf bk0, bk1;
#pragma unroll
        for (int e = 0; e < 16; e++) { bk0[e] = krow[e]; bk1[e] = krow[32 + e]; }
        sc = __builtin_amdgcn_wmma_f32_16x16x32_bf16(false, aq0, false, bk0,
                                                     (short)0, sc, false, false);
        sc = __builtin_amdgcn_wmma_f32_16x16x32_bf16(false, aq1, false, bk1,
                                                     (short)0, sc, false, false);
        float biasv = mask[b * Tsz + ((sk0 + n) & (Tsz - 1))] ? -1.0e9f : 0.0f;
#pragma unroll
        for (int r = 0; r < 8; r++) sS[(hi << 3) + r][wave * 16 + n] = sc[r] + biasv;
        __syncthreads();
        // ---- online softmax stats (parallel over 16 rows x 8 segments) ----
        {
            float pmax = -3.0e38f;
#pragma unroll
            for (int cc = 0; cc < 8; cc++) pmax = fmaxf(pmax, sS[row8][seg * 8 + cc]);
            sRed[row8][seg] = pmax;
        }
        __syncthreads();
        if (tid < 16) {
            float mo = rowM[tid];
            float mn = mo;
#pragma unroll
            for (int i = 0; i < 8; i++) mn = fmaxf(mn, sRed[tid][i]);
            rowFac[tid] = expf(mo - mn);
            rowM[tid] = mn;
        }
        __syncthreads();
        {
            float mn = rowM[row8];
            float psum = 0.0f;
#pragma unroll
            for (int cc = 0; cc < 8; cc++) {
                float p = expf(sS[row8][seg * 8 + cc] - mn);
                sP[row8][seg * 8 + cc] = f2bf(p);
                psum += p;
            }
            sRed[row8][seg] = psum;
        }
        __syncthreads();
        if (tid < 16) {
            float sum = 0.0f;
#pragma unroll
            for (int i = 0; i < 8; i++) sum += sRed[tid][i];
            rowL[tid] = rowL[tid] * rowFac[tid] + sum;
        }
        __syncthreads();
        // ---- rescale accumulator, then PV GEMM over the 64 new columns ----
#pragma unroll
        for (int r = 0; r < 8; r++) accO[r] *= rowFac[(hi << 3) + r];
#pragma unroll
        for (int half = 0; half < 2; half++) {
            int k0p = half * 32;
            v16bf ap, bv_;
#pragma unroll
            for (int e = 0; e < 16; e++) {
                int ka = kbaseB + e;
                ap[e]  = sP[mrow][k0p + ka];
                bv_[e] = vrow[chunk * 64 + k0p + kbaseB + e];   // contiguous
            }
            accO = __builtin_amdgcn_wmma_f32_16x16x32_bf16(false, ap, false, bv_,
                                                           (short)0, accO, false, false);
        }
    }
    __syncthreads();
#pragma unroll
    for (int r = 0; r < 8; r++) {
        int row = (hi << 3) + r;
        float val = accO[r] / rowL[row];
        ob[((size_t)b * Ssz + sq0 + row) * Osz + h * Dsz + dwave + n] = f2bf(val);
    }
}

// ---------------------------------------------------------------------------
// Output projection: out[b,c,s] = sum_o ob[b,s,o] * out_w[c,o] + out_b[c]
// 256-thr blocks = 8 waves = 8 16x16 tiles.
// ---------------------------------------------------------------------------
__global__ void k_outproj(const __bf16* __restrict__ A, const __bf16* __restrict__ W,
                          const float* __restrict__ bias, float* __restrict__ out) {
    int tile = blockIdx.x * 8 + (threadIdx.x >> 5);
    int ct = tile & 15;  tile >>= 4;   // C/16 = 16 tiles
    int st = tile & 127; tile >>= 7;   // S/16 = 128 tiles
    int b  = tile;
    int lane = threadIdx.x & 31;
    int hi = lane >> 4;
    int khalfA = hi << 3;
    int kbaseB = hi << 4;
    int mrow = lane & 15;
    int n = lane & 15;
    int s0 = st * 16, c0 = ct * 16;
    v8f acc = {};
    const __bf16* arow = A + ((size_t)b * Ssz + s0 + mrow) * Osz;
    const __bf16* brow = W + (size_t)(c0 + n) * Osz;
    for (int k0 = 0; k0 < Osz; k0 += 32) {
        v16bf a, bb;
#pragma unroll
        for (int e = 0; e < 16; e++) {
            int ka = kbaseB + e;
            a[e]  = arow[k0 + ka];
            bb[e] = brow[k0 + kbaseB + e];
        }
        acc = __builtin_amdgcn_wmma_f32_16x16x32_bf16(false, a, false, bb,
                                                      (short)0, acc, false, false);
    }
    float bv = bias[c0 + n];
#pragma unroll
    for (int r = 0; r < 8; r++) {
        int row = (hi << 3) + r;
        out[((size_t)b * Csz + c0 + n) * Ssz + s0 + row] = acc[r] + bv;
    }
}

// ---------------------------------------------------------------------------
extern "C" void kernel_launch(void* const* d_in, const int* in_sizes, int n_in,
                              void* d_out, int out_size, void* d_ws, size_t ws_size,
                              hipStream_t stream) {
    const float* x       = (const float*)d_in[0];
    const unsigned char* mask = (const unsigned char*)d_in[1];  // bool array, 1B/elem
    const float* q_dw_w  = (const float*)d_in[2];
    const float* q_dw_b  = (const float*)d_in[3];
    const float* q_pw_w  = (const float*)d_in[4];
    const float* q_pw_b  = (const float*)d_in[5];
    const float* k_dw_w  = (const float*)d_in[6];
    const float* k_dw_b  = (const float*)d_in[7];
    const float* k_pw_w  = (const float*)d_in[8];
    const float* k_pw_b  = (const float*)d_in[9];
    const float* v_dw_w  = (const float*)d_in[10];
    const float* v_dw_b  = (const float*)d_in[11];
    const float* v_pw_w  = (const float*)d_in[12];
    const float* v_pw_b  = (const float*)d_in[13];
    const float* out_w   = (const float*)d_in[14];
    const float* out_b   = (const float*)d_in[15];
    float* out = (float*)d_out;

    char* ws = (char*)d_ws;
    size_t off = 0;
    auto alloc = [&](size_t bytes) -> char* {
        char* p = ws + off;
        off += (bytes + 255) & ~(size_t)255;
        return p;
    };
    const size_t NW = (size_t)Osz * Csz;            // 512*256 per pw weight
    float*  cosT  = (float*)alloc((size_t)Ssz * 32 * 4);
    float*  sinT  = (float*)alloc((size_t)Ssz * 32 * 4);
    __bf16* Wq    = (__bf16*)alloc(NW * 2);
    __bf16* Wk    = (__bf16*)alloc(NW * 2);
    __bf16* Wv    = (__bf16*)alloc(NW * 2);
    __bf16* Wo    = (__bf16*)alloc(NW * 2);
    const size_t NDW = (size_t)Bsz * Ssz * Csz;     // per-branch dw activations
    __bf16* dwQ   = (__bf16*)alloc(NDW * 2);
    __bf16* dwK   = (__bf16*)alloc(NDW * 2);
    __bf16* dwV   = (__bf16*)alloc(NDW * 2);
    const size_t NY = (size_t)Bsz * Ssz * Osz;      // per-branch pw output (fp32)
    float*  Y     = (float*)alloc(3 * NY * 4);      // [q | k | v]
    const size_t NQ = (size_t)Bsz * Hsz * Ssz * Dsz;
    __bf16* qb    = (__bf16*)alloc(NQ * 2);
    __bf16* kb    = (__bf16*)alloc(NQ * 2);
    __bf16* vt    = (__bf16*)alloc(NQ * 2);         // V transposed [b][h][d][s]
    __bf16* ob    = (__bf16*)alloc(NY * 2);         // attention output, bf16 [b][s][512]
    (void)ws_size; (void)in_sizes; (void)n_in; (void)out_size;

    // 1. tables + weight conversion
    k_rope_table<<<(Ssz * 32 + 255) / 256, 256, 0, stream>>>(cosT, sinT);
    k_cvt_bf16<<<(int)((NW + 255) / 256), 256, 0, stream>>>(q_pw_w, Wq, (int)NW);
    k_cvt_bf16<<<(int)((NW + 255) / 256), 256, 0, stream>>>(k_pw_w, Wk, (int)NW);
    k_cvt_bf16<<<(int)((NW + 255) / 256), 256, 0, stream>>>(v_pw_w, Wv, (int)NW);
    k_cvt_bf16<<<(int)((NW + 255) / 256), 256, 0, stream>>>(out_w, Wo, (int)NW);

    // 2. depthwise convs
    k_dwconv<<<Bsz * Csz, 256, 0, stream>>>(x, q_dw_w, q_dw_b, dwQ);
    k_dwconv<<<Bsz * Csz, 256, 0, stream>>>(x, k_dw_w, k_dw_b, dwK);
    k_dwconv<<<Bsz * Csz, 256, 0, stream>>>(x, v_dw_w, v_dw_b, dwV);

    // 3. pointwise GEMMs (WMMA), 8 tiles per 256-thread block
    const int pwTiles = Bsz * (Ssz / 16) * (Osz / 16);   // 16384 tiles
    k_pwgemm<<<pwTiles / 8, 256, 0, stream>>>(dwQ, Wq, q_pw_b, Y);
    k_pwgemm<<<pwTiles / 8, 256, 0, stream>>>(dwK, Wk, k_pw_b, Y + NY);
    k_pwgemm<<<pwTiles / 8, 256, 0, stream>>>(dwV, Wv, v_pw_b, Y + 2 * NY);

    // 4. RoPE + relayout + bf16 (V transposed)
    k_rope_apply<<<(Bsz * Hsz * Ssz * 32) / 256, 256, 0, stream>>>(Y, cosT, sinT, qb, kb, vt);

    // 5. flash attention (WMMA)
    k_attn<<<Bsz * Hsz * (Ssz / 16), 128, 0, stream>>>(qb, kb, vt, mask, ob);

    // 6. output projection (WMMA), 8 tiles per 256-thread block
    k_outproj<<<(Bsz * (Ssz / 16) * (Csz / 16)) / 8, 256, 0, stream>>>(ob, Wo, out_b, out);
}
